// ObjectQueryMaskedMHCA_27169963115069
// MI455X (gfx1250) — matrix operand order repaired and checked by reading, not compiled, not run on hardware
//
#include <hip/hip_runtime.h>
#include <hip/hip_bf16.h>

// ---------------------------------------------------------------------------
// Problem constants (from reference)
// ---------------------------------------------------------------------------
constexpr int NB  = 2;      // B
constexpr int NO  = 8;      // O
constexpr int NCH = 1024;   // C
constexpr int NT  = 1024;   // T
constexpr int NH  = 16;     // heads
constexpr int NDC = 64;     // NC = C/H
constexpr int NW  = 24;     // O*WIN key/value streams per batch

typedef __attribute__((ext_vector_type(16))) __bf16 v16bf;
typedef __attribute__((ext_vector_type(8)))  float  v8f;

// ---------------------------------------------------------------------------
// fp32 -> bf16 conversion (weights)
// ---------------------------------------------------------------------------
__global__ void f2bf_kernel(const float* __restrict__ in, __bf16* __restrict__ out, int n) {
    int i = blockIdx.x * 256 + threadIdx.x;
    if (i < n) out[i] = (__bf16)in[i];
}

// ---------------------------------------------------------------------------
// Stage 1: depthwise conv3 (+window shift) * mask, then channel LayerNorm,
// store bf16 plane (P, C, T).
// block = (32 lanes over t, 8 rows over channels), grid = (T/32, planes)
// ---------------------------------------------------------------------------
__device__ __forceinline__ float conv3_val(const float* __restrict__ xp,
                                           const float* __restrict__ cw,
                                           int c, int t, int shift) {
    float acc = 0.f;
#pragma unroll
    for (int d = -1; d <= 1; ++d) {
        int ts  = t + d;          // tap position in the shifted signal
        int txi = ts + shift;     // underlying x index
        if (ts >= 0 && ts < NT && txi >= 0 && txi < NT)
            acc += cw[c * 3 + (d + 1)] * xp[(size_t)c * NT + txi];
    }
    return acc;
}

__global__ void __launch_bounds__(256)
prep_kernel(const float* __restrict__ x,            // (B,O,C,T)
            const float* __restrict__ cw,           // (C,1,3)
            const float* __restrict__ g,            // (C,)
            const float* __restrict__ be,           // (C,)
            const unsigned char* __restrict__ mask, // (B,O,T) bool
            __bf16* __restrict__ out,               // (P,C,T)
            int windowed)                           // 0: 16 planes, 1: 48 planes
{
    const int t = blockIdx.x * 32 + threadIdx.x;
    const int p = blockIdx.y;
    int b, o, shift;
    if (windowed) { b = p / 24; int r = p % 24; o = r / 3; shift = (r % 3) - 1; }
    else          { b = p / NO; o = p % NO; shift = 0; }

    const float* xp = x + (size_t)(b * NO + o) * NCH * NT;
    const int tm = t + shift;
    const float mw = (tm >= 0 && tm < NT)
                   ? (float)mask[(size_t)(b * NO + o) * NT + tm] : 0.f;

    __shared__ float s_sum[8][32];
    __shared__ float s_sq[8][32];
    __shared__ float s_mu[32];
    __shared__ float s_rs[32];

    float sum = 0.f, sq = 0.f;
    for (int c = threadIdx.y; c < NCH; c += 8) {
        float a = conv3_val(xp, cw, c, t, shift) * mw;
        sum += a; sq += a * a;
    }
    s_sum[threadIdx.y][threadIdx.x] = sum;
    s_sq [threadIdx.y][threadIdx.x] = sq;
    __syncthreads();
    if (threadIdx.y == 0) {
        float s = 0.f, q2 = 0.f;
#pragma unroll
        for (int i = 0; i < 8; ++i) { s += s_sum[i][threadIdx.x]; q2 += s_sq[i][threadIdx.x]; }
        float mu  = s / (float)NCH;
        float var = q2 / (float)NCH - mu * mu;
        s_mu[threadIdx.x] = mu;
        s_rs[threadIdx.x] = rsqrtf(var + 1e-5f);
    }
    __syncthreads();
    const float mu = s_mu[threadIdx.x];
    const float rs = s_rs[threadIdx.x];
    for (int c = threadIdx.y; c < NCH; c += 8) {
        float a = conv3_val(xp, cw, c, t, shift) * mw;
        float y = (a - mu) * rs * g[c] + be[c];
        out[((size_t)p * NCH + c) * NT + t] = (__bf16)y;
    }
}

// ---------------------------------------------------------------------------
// Stage 2/4: bf16 GEMM  Y[p] = W(C x C) @ X[p](C x T), via v_wmma_f32_16x16x32_bf16
// Block tile 128(M) x 128(N), K-step 32.
//  - W tile staged global->LDS with GLOBAL_LOAD_ASYNC_TO_LDS_B128 (ASYNCcnt),
//    no VGPR round trip; row-major layout matches the A fragment directly.
//  - X tile loaded b128 to VGPRs and transpose-stored into sXT[N][K] so B
//    fragments are two contiguous ds_load_b128 per tile (no scalar gathers).
// 8 waves: 2 (M) x 4 (N); each wave owns a 64x32 tile = 4x2 WMMA fragments.
// Epilogue: (+bias) * scale, optional per-(plane,t) mask, bf16 or f32 out.
// ---------------------------------------------------------------------------
#define TM 128
#define TN 128
#define TK 32

__global__ void __launch_bounds__(256)
gemm_bf16_kernel(const __bf16* __restrict__ W,            // (C, C) row-major
                 const __bf16* __restrict__ X,            // (P, C, T)
                 const float*  __restrict__ bias,         // (C,)
                 float scale,
                 const unsigned char* __restrict__ mask,  // (P, T) or null
                 __bf16* __restrict__ outB,               // (P, C, T) or null
                 float*  __restrict__ outF)               // (P, C, T) or null
{
    __shared__ __bf16 sW [TM][TK + 8];   // row-major: [m][k]
    __shared__ __bf16 sXT[TN][TK + 8];   // transposed: [n][k]

    const int tid  = threadIdx.x;
    const int lane = tid & 31;
    const int wave = tid >> 5;
    const int wm   = wave >> 2;   // 0..1  (64 rows each)
    const int wn   = wave & 3;    // 0..3  (32 cols each)
    const int n0   = blockIdx.x * TN;
    const int m0   = blockIdx.y * TM;
    const int p    = blockIdx.z;
    const __bf16* Xp = X + (size_t)p * NCH * NT;

    const int hh   = lane >> 4;   // lane half (0/1)
    const int l15  = lane & 15;

    v8f acc[4][2] = {};

    const int lwr = tid >> 1;            // W-tile row      0..127
    const int lwc = (tid & 1) * 16;      // W-tile col base 0/16
    const int lxr = tid >> 3;            // X-tile k row    0..31
    const int lxc = (tid & 7) * 16;      // X-tile n base

    for (int k0 = 0; k0 < NCH; k0 += TK) {
        // --- W tile: async global -> LDS (b128 x2 per thread, ASYNCcnt) ---
        {
            const __bf16* wsrc = W + (size_t)(m0 + lwr) * NCH + k0 + lwc;
            unsigned lds_off = (unsigned)(uintptr_t)(void*)&sW[lwr][lwc];
            unsigned long long ga = (unsigned long long)(uintptr_t)wsrc;
            asm volatile("global_load_async_to_lds_b128 %0, %1, off"
                         :: "v"(lds_off), "v"(ga) : "memory");
            asm volatile("global_load_async_to_lds_b128 %0, %1, off offset:16"
                         :: "v"(lds_off), "v"(ga) : "memory");
        }
        // --- X tile: b128 loads, transpose-store into sXT[n][k] ---
        {
            const __bf16* xsrc = Xp + (size_t)(k0 + lxr) * NT + n0 + lxc;
            uint4 d0 = *(const uint4*)(xsrc);
            uint4 d1 = *(const uint4*)(xsrc + 8);
            const __bf16* e0 = (const __bf16*)&d0;
            const __bf16* e1 = (const __bf16*)&d1;
#pragma unroll
            for (int i = 0; i < 8; ++i) sXT[lxc + i][lxr]     = e0[i];
#pragma unroll
            for (int i = 0; i < 8; ++i) sXT[lxc + 8 + i][lxr] = e1[i];
            // hint next K tile toward the caches (global_prefetch_b8)
            __builtin_prefetch(xsrc + (size_t)TK * NT, 0, 1);
        }
        // all async LDS writes from this wave must land before the barrier
        asm volatile("s_wait_asynccnt 0x0" ::: "memory");
        __syncthreads();

        // --- fragments per ISA layouts (contiguous b128 LDS reads) ---
        // A (16x32 bf16): row = lane&15; kk = 8*hh + 0..7, then +16
        v16bf a[4];
#pragma unroll
        for (int mt = 0; mt < 4; ++mt) {
            const __bf16* ap = &sW[wm * 64 + mt * 16 + l15][hh * 8];
#pragma unroll
            for (int j = 0; j < 8; ++j) a[mt][j]     = ap[j];
#pragma unroll
            for (int j = 0; j < 8; ++j) a[mt][8 + j] = ap[16 + j];
        }
        // B (32x16 bf16): N = lane&15; K = 8*hh + 0..7, then +16 (matches A)
        v16bf bfr[2];
#pragma unroll
        for (int nt = 0; nt < 2; ++nt) {
            const __bf16* bp = &sXT[wn * 32 + nt * 16 + l15][hh * 8];
#pragma unroll
            for (int j = 0; j < 16; ++j) bfr[nt][j] = bp[(j & 8) * 2 + (j & 7)];
        }

#pragma unroll
        for (int mt = 0; mt < 4; ++mt)
#pragma unroll
            for (int nt = 0; nt < 2; ++nt)
                acc[mt][nt] = __builtin_amdgcn_wmma_f32_16x16x32_bf16(
                    false, a[mt], false, bfr[nt], (short)0, acc[mt][nt], false, false);
        __syncthreads();
    }

    // --- epilogue: D layout VGPR j -> M = j + 8*hh, N = lane&15 ---
#pragma unroll
    for (int mt = 0; mt < 4; ++mt) {
#pragma unroll
        for (int nt = 0; nt < 2; ++nt) {
            const int colg = n0 + wn * 32 + nt * 16 + l15;
            const float mv = mask ? (float)mask[(size_t)p * NT + colg] : 1.f;
#pragma unroll
            for (int j = 0; j < 8; ++j) {
                const int rowg = m0 + wm * 64 + mt * 16 + j + 8 * hh;
                float v = (acc[mt][nt][j] + bias[rowg]) * scale * mv;
                const size_t idx = ((size_t)p * NCH + rowg) * NT + colg;
                if (outF) outF[idx] = v;
                else      outB[idx] = (__bf16)v;
            }
        }
    }
}

// ---------------------------------------------------------------------------
// Stage 3: per-timestep 24-way masked softmax attention over NC=64 channels.
// One lane per (b,o,h,t). q already carries the 1/sqrt(NC) scale.
// grid = (T/256, H, B*O), block = 256.
// ---------------------------------------------------------------------------
__global__ void __launch_bounds__(256)
attn_kernel(const __bf16* __restrict__ q,            // (16, C, T)
            const __bf16* __restrict__ k,            // (48, C, T)
            const __bf16* __restrict__ v,            // (48, C, T)
            const unsigned char* __restrict__ mask,  // (B,O,T)
            __bf16* __restrict__ ao)                 // (16, C, T)
{
    const int t  = blockIdx.x * 256 + threadIdx.x;
    const int h  = blockIdx.y;
    const int bo = blockIdx.z;
    const int b  = bo >> 3;
    const int c0 = h * NDC;
    const size_t plq = (size_t)bo * NCH * NT;

    float qr[NDC];
#pragma unroll
    for (int c = 0; c < NDC; ++c)
        qr[c] = (float)q[plq + (size_t)(c0 + c) * NT + t];

    float s[NW];
    float vm[NW];
    for (int w = 0; w < NW; ++w) {
        const int o2 = w / 3;
        const int sh = (w % 3) - 1;
        const int tmv = t + sh;
        vm[w] = (tmv >= 0 && tmv < NT)
              ? (float)mask[(size_t)(b * NO + o2) * NT + tmv] : 0.f;
        const __bf16* kp = k + ((size_t)(b * NW + w) * NCH + c0) * NT + t;
        float acc = 0.f;
#pragma unroll
        for (int c = 0; c < NDC; ++c) acc += qr[c] * (float)kp[(size_t)c * NT];
        s[w] = acc;
    }

    const float qm = (float)mask[(size_t)bo * NT + t];
    float mx = -1e30f;
#pragma unroll
    for (int w = 0; w < NW; ++w) {
        float z = (qm > 0.f) ? s[w] : -1e9f;
        s[w] = z;
        mx = fmaxf(mx, z);
    }
    float ssum = 0.f;
#pragma unroll
    for (int w = 0; w < NW; ++w) { s[w] = __expf(s[w] - mx); ssum += s[w]; }
    const float inv = (qm > 0.f) ? (1.f / ssum) : 0.f;
#pragma unroll
    for (int w = 0; w < NW; ++w) s[w] = s[w] * inv * vm[w];

    for (int c = 0; c < NDC; ++c) {
        float acc = 0.f;
#pragma unroll
        for (int w = 0; w < NW; ++w)
            acc += s[w] * (float)v[((size_t)(b * NW + w) * NCH + c0 + c) * NT + t];
        ao[plq + (size_t)(c0 + c) * NT + t] = (__bf16)acc;
    }
}

// ---------------------------------------------------------------------------
// Host-side orchestration
// ---------------------------------------------------------------------------
extern "C" void kernel_launch(void* const* d_in, const int* in_sizes, int n_in,
                              void* d_out, int out_size, void* d_ws, size_t ws_size,
                              hipStream_t stream) {
    (void)in_sizes; (void)n_in; (void)out_size; (void)ws_size;

    const float* x    = (const float*)d_in[0];
    const unsigned char* mask = (const unsigned char*)d_in[1]; // bool
    const float* qc_w = (const float*)d_in[2];
    const float* kc_w = (const float*)d_in[3];
    const float* vc_w = (const float*)d_in[4];
    const float* qn_w = (const float*)d_in[5];
    const float* qn_b = (const float*)d_in[6];
    const float* kn_w = (const float*)d_in[7];
    const float* kn_b = (const float*)d_in[8];
    const float* vn_w = (const float*)d_in[9];
    const float* vn_b = (const float*)d_in[10];
    const float* q_w  = (const float*)d_in[11];
    const float* q_b  = (const float*)d_in[12];
    const float* k_w  = (const float*)d_in[13];
    const float* k_b  = (const float*)d_in[14];
    const float* v_w  = (const float*)d_in[15];
    const float* v_b  = (const float*)d_in[16];
    const float* p_w  = (const float*)d_in[17];
    const float* p_b  = (const float*)d_in[18];

    const size_t CC = (size_t)NCH * NCH;
    const size_t PT = (size_t)NCH * NT;

    __bf16* ws = (__bf16*)d_ws;
    __bf16* wq = ws;                 // weights, bf16
    __bf16* wk = wq + CC;
    __bf16* wv = wk + CC;
    __bf16* wp = wv + CC;
    __bf16* qn = wp + CC;            // 16 planes
    __bf16* kn = qn + 16 * PT;       // 48 planes
    __bf16* vn = kn + 48 * PT;       // 48 planes
    __bf16* qb = vn + 48 * PT;       // 16 planes
    __bf16* kb = qb + 16 * PT;       // 48 planes
    __bf16* vb = kb + 48 * PT;       // 48 planes
    __bf16* ao = qn;                 // reuse qn (dead after q GEMM)

    const int ncc = (int)CC;
    f2bf_kernel<<<dim3((ncc + 255) / 256), 256, 0, stream>>>(q_w, wq, ncc);
    f2bf_kernel<<<dim3((ncc + 255) / 256), 256, 0, stream>>>(k_w, wk, ncc);
    f2bf_kernel<<<dim3((ncc + 255) / 256), 256, 0, stream>>>(v_w, wv, ncc);
    f2bf_kernel<<<dim3((ncc + 255) / 256), 256, 0, stream>>>(p_w, wp, ncc);

    // Stage 1: conv + mask + LN -> bf16
    prep_kernel<<<dim3(NT / 32, 16), dim3(32, 8), 0, stream>>>(x, qc_w, qn_w, qn_b, mask, qn, 0);
    prep_kernel<<<dim3(NT / 32, 48), dim3(32, 8), 0, stream>>>(x, kc_w, kn_w, kn_b, mask, kn, 1);
    prep_kernel<<<dim3(NT / 32, 48), dim3(32, 8), 0, stream>>>(x, vc_w, vn_w, vn_b, mask, vn, 1);

    // Stage 2: pointwise GEMMs (q carries 1/sqrt(NC))
    gemm_bf16_kernel<<<dim3(NT / TN, NCH / TM, 16), 256, 0, stream>>>(
        wq, qn, q_b, 0.125f, nullptr, qb, nullptr);
    gemm_bf16_kernel<<<dim3(NT / TN, NCH / TM, 48), 256, 0, stream>>>(
        wk, kn, k_b, 1.0f, nullptr, kb, nullptr);
    gemm_bf16_kernel<<<dim3(NT / TN, NCH / TM, 48), 256, 0, stream>>>(
        wv, vn, v_b, 1.0f, nullptr, vb, nullptr);

    // Stage 3: per-timestep masked softmax attention
    attn_kernel<<<dim3(NT / 256, NH, NB * NO), 256, 0, stream>>>(qb, kb, vb, mask, ao);

    // Stage 4: output projection, bias + query-mask, f32 out
    gemm_bf16_kernel<<<dim3(NT / TN, NCH / TM, 16), 256, 0, stream>>>(
        wp, ao, p_b, 1.0f, mask, nullptr, (float*)d_out);
}
